// CollaborativeAttention_21663815041721
// MI455X (gfx1250) — matrix operand order repaired and checked by reading, not compiled, not run on hardware
//
#include <hip/hip_runtime.h>
#include <hip/hip_bf16.h>

typedef __attribute__((ext_vector_type(16))) __bf16 bf16x16;
typedef __attribute__((ext_vector_type(8)))  float  floatx8;

#define WMMA_BF16(A, B, C) \
    __builtin_amdgcn_wmma_f32_16x16x32_bf16(false, (A), false, (B), (short)0, (C), false, false)

static constexpr int Bn   = 2;
static constexpr int Sn   = 2048;
static constexpr int Dn   = 1024;   // DIN = DK = DV = DOUT
static constexpr int Hn   = 16;
static constexpr int Dh   = 64;     // head size
static constexpr float SCALE = 0.125f;  // 1/sqrt(64)

union FragU { uint4 q[2]; bf16x16 v; };

static __device__ inline bf16x16 ldfrag(const __bf16* base, int e0, int e1) {
    FragU u;
    u.q[0] = *(const uint4*)(base + e0);
    u.q[1] = *(const uint4*)(base + e1);
    return u.v;
}

// ---- CDNA5 async global->LDS DMA (ASYNCcnt-tracked), 16B per lane ----
static __device__ inline void async_b128(const void* lds_ptr, const void* gptr) {
    asm volatile("global_load_async_to_lds_b128 %0, %1, off"
                 :: "v"((unsigned)(uintptr_t)lds_ptr),
                    "v"((unsigned long long)(uintptr_t)gptr)
                 : "memory");
}
static __device__ inline void wait_async_le4()  { asm volatile("s_wait_asynccnt 4"  ::: "memory"); }
static __device__ inline void wait_async_le8()  { asm volatile("s_wait_asynccnt 8"  ::: "memory"); }

// ---------------------------------------------------------------- cvt f32->bf16
__global__ __launch_bounds__(256) void cvt_bf16_kernel(const float* __restrict__ src,
                                                       __bf16* __restrict__ dst, int n) {
    int i = blockIdx.x * 256 + threadIdx.x;
    if (i < n) dst[i] = (__bf16)src[i];
}

// ---------------------------------------------------------------- generic GEMM
// C[m,n] = sum_k A[m,k] * Bw[n,k]  (+bias[n]) (+resid[m*N+n])
// A: [M,K] bf16 row-major, Bw: [N,K] bf16 row-major. M % 128 == 0, K % 64 == 0.
// Triple-buffered 64-wide K stages, async-DMA prefetch, one barrier per stage.
template<bool OUT_F32, bool OUT_BF16, bool BIAS, bool RESID>
__global__ __launch_bounds__(256)
void gemm_nt(const __bf16* __restrict__ A, const __bf16* __restrict__ Bw,
             int M, int N, int K,
             const float* __restrict__ bias, const float* __restrict__ resid,
             float* __restrict__ Cf, __bf16* __restrict__ Cb) {
    __shared__ __bf16 As[3][128][72];   // [buf][m][k], padded
    __shared__ __bf16 Bs[3][128][72];   // [buf][n][k], padded
    const int tid  = threadIdx.x;
    const int lane = tid & 31;
    const int w    = tid >> 5;
    const int m0   = blockIdx.x * 128;
    const int n0   = blockIdx.y * 128;
    const int wm   = (w >> 2) * 64;      // 2 waves along M
    const int wn   = (w & 3) * 32;       // 4 waves along N

    floatx8 acc[4][2];
    for (int i = 0; i < 4; ++i) for (int j = 0; j < 2; ++j) acc[i][j] = (floatx8)0.0f;

    const int lrow = tid >> 1;            // 0..127
    const int c0   = (tid & 1) * 32;      // 32 bf16 = 4 x 16B per thread per matrix
    const int brow = (n0 + lrow < N) ? (n0 + lrow) : (N - 1);  // clamp (cols>=N never stored)
    const __bf16* gA0 = A  + (size_t)(m0 + lrow) * K + c0;
    const __bf16* gB0 = Bw + (size_t)brow        * K + c0;

    auto issue = [&](int kk, int buf) {   // stage at K offset kk -> LDS buffer buf
        const __bf16* ga = gA0 + kk;
        const __bf16* gb = gB0 + kk;
        const __bf16* la = &As[buf][lrow][c0];
        const __bf16* lb = &Bs[buf][lrow][c0];
        #pragma unroll
        for (int c = 0; c < 4; ++c) {
            async_b128(la + c * 8, ga + c * 8);
            async_b128(lb + c * 8, gb + c * 8);
        }
    };

    const int am = lane & 15, ah = (lane >> 4) * 8;
    const int bn = lane & 15, bh = (lane >> 4) * 8;

    issue(0, 0);
    int kk_next = 64;        // coords of next prefetch
    int bc = 0, bx = 1;      // rotating buffer indices: compute / next-issue
    for (int kk = 0; kk < K; kk += 64) {
        issue(kk_next, bx);              // prefetch next stage
        kk_next += 64; if (kk_next >= K) kk_next = 0;
        wait_async_le8();                // our batch for this stage has retired
        __syncthreads();                 // everyone's stage data visible
        const __bf16 (*Ab)[72] = As[bc];
        const __bf16 (*Bb)[72] = Bs[bc];
        #pragma unroll
        for (int ksub = 0; ksub < 2; ++ksub) {
            bf16x16 afr[4], bfr[2];
            #pragma unroll
            for (int fm = 0; fm < 4; ++fm)
                afr[fm] = ldfrag(&Ab[wm + fm * 16 + am][0],
                                 ksub * 32 + ah, ksub * 32 + 16 + ah);
            #pragma unroll
            for (int fn = 0; fn < 2; ++fn)
                bfr[fn] = ldfrag(&Bb[wn + fn * 16 + bn][0],
                                 ksub * 32 + bh, ksub * 32 + 16 + bh);
            #pragma unroll
            for (int fm = 0; fm < 4; ++fm)
                #pragma unroll
                for (int fn = 0; fn < 2; ++fn)
                    acc[fm][fn] = WMMA_BF16(afr[fm], bfr[fn], acc[fm][fn]);
        }
        bc = (bc == 2) ? 0 : bc + 1;
        bx = (bx == 2) ? 0 : bx + 1;
    }

    // store: C frag row m = r + (lane>=16)*8, col n = lane&15
    const int nl = lane & 15;
    const int mb = (lane >> 4) * 8;
    for (int fm = 0; fm < 4; ++fm)
        for (int fn = 0; fn < 2; ++fn)
            for (int r = 0; r < 8; ++r) {
                int m = m0 + wm + fm * 16 + mb + r;
                int n = n0 + wn + fn * 16 + nl;
                if (n < N) {
                    float v = acc[fm][fn][r];
                    if (BIAS)  v += bias[n];
                    if (RESID) v += resid[(size_t)m * N + n];
                    if (OUT_F32)  Cf[(size_t)m * N + n] = v;
                    if (OUT_BF16) Cb[(size_t)m * N + n] = (__bf16)v;
                }
            }
}

// ---------------------------------------------------------------- flash attention
// grid: (S/32, H, B); block: 128 threads (4 waves).
// scores[i,j] = (sum_c (q[i,c]*mixing[h,c]) * k[j,c] + cb[j,h]) * SCALE
__global__ __launch_bounds__(128)
void attn_kernel(const float* __restrict__ qf, const __bf16* __restrict__ kbf,
                 const __bf16* __restrict__ vbf, const float* __restrict__ cb,
                 const float* __restrict__ mixing, __bf16* __restrict__ ctx) {
    __shared__ __bf16 qs[32][1032];      // q*mixing, bf16, resident (66 KB)
    __shared__ __bf16 ks[3][64][72];     // k stage buffers [j][k] (27 KB)
    __shared__ float  ss[32][68];        // score tile
    __shared__ __bf16 ps[32][72];        // softmax probs (bf16)
    __shared__ __bf16 vt[64][72];        // v tile transposed [c][j]
    __shared__ float  mrow[32], lrow[32], crow[32];

    const int tid = threadIdx.x, lane = tid & 31, w = tid >> 5;
    const int i0 = blockIdx.x * 32, h = blockIdx.y, b = blockIdx.z;
    const size_t base = (size_t)b * Sn;

    const int krow = tid >> 1;            // 0..63
    const int kc0  = (tid & 1) * 32;      // 32 bf16 = 4 x 16B per thread
    const __bf16* gk0 = kbf + (base + krow) * Dn + kc0;   // + j*Dn + k

    auto issue_ks = [&](int jp, int kp, int buf) {
        const __bf16* g = gk0 + (size_t)jp * Dn + kp;
        const __bf16* l = &ks[buf][krow][kc0];
        #pragma unroll
        for (int c = 0; c < 4; ++c) async_b128(l + c * 8, g + c * 8);
    };

    issue_ks(0, 0, 0);   // prefetch first k stage, overlaps with q-residency setup

    // load q rows and fold in mixing[h] (fp32), convert to bf16
    {
        const int row = tid >> 2, c0 = (tid & 3) * 256;
        const float* qrow = qf + (base + i0 + row) * Dn + c0;
        const float* mx   = mixing + h * Dn + c0;
        for (int c = 0; c < 256; c += 4) {
            float4 qv = *(const float4*)(qrow + c);
            float4 mv = *(const float4*)(mx + c);
            union { __bf16 e[4]; uint2 u; } pk;
            pk.e[0] = (__bf16)(qv.x * mv.x);
            pk.e[1] = (__bf16)(qv.y * mv.y);
            pk.e[2] = (__bf16)(qv.z * mv.z);
            pk.e[3] = (__bf16)(qv.w * mv.w);
            *(uint2*)(&qs[row][c0 + c]) = pk.u;
        }
    }
    if (tid < 32) { mrow[tid] = -1e30f; lrow[tid] = 0.0f; }
    floatx8 cacc[2];
    cacc[0] = (floatx8)0.0f; cacc[1] = (floatx8)0.0f;

    const int am = lane & 15, ah = (lane >> 4) * 8;
    const int nl = lane & 15, bh = (lane >> 4) * 8;
    const int mb = (lane >> 4) * 8;

    int jn = 0, kn = 64;   // coords of next prefetch (carried, no decode)
    int bc = 0, bx = 1;    // rotating buffer indices
    for (int j0 = 0; j0 < Sn; j0 += 64) {
        // ---- scores: sacc = qm @ k^T over K=1024, pipelined in 64-wide stages
        floatx8 sacc[2];
        sacc[0] = (floatx8)0.0f; sacc[1] = (floatx8)0.0f;
        for (int kk = 0; kk < Dn; kk += 64) {
            issue_ks(jn, kn, bx);        // prefetch next stage
            kn += 64;
            if (kn >= Dn) { kn = 0; jn += 64; if (jn >= Sn) jn = 0; }
            wait_async_le4();
            __syncthreads();
            const __bf16 (*kb)[72] = ks[bc];
            #pragma unroll
            for (int ksub = 0; ksub < 2; ++ksub) {
                bf16x16 bfr = ldfrag(&kb[w * 16 + nl][0],
                                     ksub * 32 + bh, ksub * 32 + 16 + bh);
                #pragma unroll
                for (int fm = 0; fm < 2; ++fm) {
                    bf16x16 afr = ldfrag(&qs[fm * 16 + am][0],
                                         kk + ksub * 32 + ah, kk + ksub * 32 + 16 + ah);
                    sacc[fm] = WMMA_BF16(afr, bfr, sacc[fm]);
                }
            }
            bc = (bc == 2) ? 0 : bc + 1;
            bx = (bx == 2) ? 0 : bx + 1;
        }
        // ---- write scaled+biased scores to LDS
        {
            const int jg = j0 + w * 16 + nl;
            const float cbv = cb[(base + jg) * Hn + h];
            for (int fm = 0; fm < 2; ++fm)
                for (int r = 0; r < 8; ++r)
                    ss[fm * 16 + mb + r][w * 16 + nl] = (sacc[fm][r] + cbv) * SCALE;
        }
        __syncthreads();
        // ---- online softmax (32 rows) + v-tile transposed load (all threads)
        {
            const int j = tid >> 1, c0 = (tid & 1) * 32;
            for (int c = 0; c < 32; c += 8) {
                union { uint4 u; __bf16 e[8]; } uu;
                uu.u = *(const uint4*)(vbf + (base + j0 + j) * Dn + h * Dh + c0 + c);
                for (int e = 0; e < 8; ++e) vt[c0 + c + e][j] = uu.e[e];
            }
        }
        if (tid < 32) {
            const int r = tid;
            float mold = mrow[r];
            float tm = -1e30f;
            for (int j = 0; j < 64; ++j) tm = fmaxf(tm, ss[r][j]);
            float mnew = fmaxf(mold, tm);
            float corr = __expf(mold - mnew);
            float sum = 0.0f;
            for (int j = 0; j < 64; ++j) {
                float p = __expf(ss[r][j] - mnew);
                ps[r][j] = (__bf16)p;
                sum += p;
            }
            lrow[r] = lrow[r] * corr + sum;
            mrow[r] = mnew;
            crow[r] = corr;
        }
        __syncthreads();
        // ---- rescale ctx acc, then ctx += P @ V  (K=64 in two 32-steps)
        for (int fm = 0; fm < 2; ++fm)
            for (int r = 0; r < 8; ++r)
                cacc[fm][r] *= crow[fm * 16 + mb + r];
        #pragma unroll
        for (int k2 = 0; k2 < 2; ++k2) {
            bf16x16 bfr = ldfrag(&vt[w * 16 + nl][0], k2 * 32 + bh, k2 * 32 + 16 + bh);
            #pragma unroll
            for (int fm = 0; fm < 2; ++fm) {
                bf16x16 afr = ldfrag(&ps[fm * 16 + am][0], k2 * 32 + ah, k2 * 32 + 16 + ah);
                cacc[fm] = WMMA_BF16(afr, bfr, cacc[fm]);
            }
        }
    }

    // ---- epilogue: ctx / l  -> bf16
    for (int fm = 0; fm < 2; ++fm)
        for (int r = 0; r < 8; ++r) {
            int m = fm * 16 + mb + r;
            float val = cacc[fm][r] / lrow[m];
            ctx[(base + i0 + m) * Dn + h * Dh + w * 16 + nl] = (__bf16)val;
        }
}

// ---------------------------------------------------------------- layernorm
__global__ __launch_bounds__(256)
void ln_kernel(const float* __restrict__ res, const float* __restrict__ gamma,
               const float* __restrict__ beta, float* __restrict__ out) {
    __shared__ float red[256];
    const int row = blockIdx.x, tid = threadIdx.x;
    const float* r = res + (size_t)row * Dn;
    float x[4], s = 0.0f;
    for (int i = 0; i < 4; ++i) { x[i] = r[tid + i * 256]; s += x[i]; }
    red[tid] = s; __syncthreads();
    for (int off = 128; off > 0; off >>= 1) {
        if (tid < off) red[tid] += red[tid + off];
        __syncthreads();
    }
    const float mu = red[0] / (float)Dn;
    __syncthreads();
    float v = 0.0f;
    for (int i = 0; i < 4; ++i) { float d = x[i] - mu; v += d * d; }
    red[tid] = v; __syncthreads();
    for (int off = 128; off > 0; off >>= 1) {
        if (tid < off) red[tid] += red[tid + off];
        __syncthreads();
    }
    const float inv = rsqrtf(red[0] / (float)Dn + 1e-5f);
    for (int i = 0; i < 4; ++i) {
        int c = tid + i * 256;
        out[(size_t)row * Dn + c] = (x[i] - mu) * inv * gamma[c] + beta[c];
    }
}

// ---------------------------------------------------------------- launch
extern "C" void kernel_launch(void* const* d_in, const int* in_sizes, int n_in,
                              void* d_out, int out_size, void* d_ws, size_t ws_size,
                              hipStream_t stream) {
    const float* x      = (const float*)d_in[0];
    const float* Wq     = (const float*)d_in[1];
    const float* Wk     = (const float*)d_in[2];
    const float* Wcb    = (const float*)d_in[3];
    const float* Wv     = (const float*)d_in[4];
    const float* bv     = (const float*)d_in[5];
    const float* mixing = (const float*)d_in[6];
    const float* Wd     = (const float*)d_in[7];
    const float* bd     = (const float*)d_in[8];
    const float* gamma  = (const float*)d_in[9];
    const float* beta   = (const float*)d_in[10];
    float* out = (float*)d_out;

    const int M = Bn * Sn;              // 4096
    char* p = (char*)d_ws;
    auto alloc = [&](size_t bytes) { void* r = (void*)p; p += (bytes + 255) & ~(size_t)255; return r; };
    __bf16* x_bf   = (__bf16*)alloc((size_t)M * Dn * 2);
    __bf16* wq_bf  = (__bf16*)alloc((size_t)Dn * Dn * 2);
    __bf16* wk_bf  = (__bf16*)alloc((size_t)Dn * Dn * 2);
    __bf16* wv_bf  = (__bf16*)alloc((size_t)Dn * Dn * 2);
    __bf16* wd_bf  = (__bf16*)alloc((size_t)Dn * Dn * 2);
    __bf16* wcb_bf = (__bf16*)alloc((size_t)Hn * Dn * 2);
    float*  q_f    = (float*) alloc((size_t)M * Dn * 4);
    __bf16* k_bf   = (__bf16*)alloc((size_t)M * Dn * 2);
    __bf16* v_bf   = (__bf16*)alloc((size_t)M * Dn * 2);
    float*  cb_f   = (float*) alloc((size_t)M * Hn * 4);
    __bf16* ctx_bf = (__bf16*)alloc((size_t)M * Dn * 2);
    float*  res_f  = (float*) alloc((size_t)M * Dn * 4);

    // 1) bf16 conversions
    cvt_bf16_kernel<<<(M * Dn + 255) / 256, 256, 0, stream>>>(x, x_bf, M * Dn);
    cvt_bf16_kernel<<<(Dn * Dn + 255) / 256, 256, 0, stream>>>(Wq, wq_bf, Dn * Dn);
    cvt_bf16_kernel<<<(Dn * Dn + 255) / 256, 256, 0, stream>>>(Wk, wk_bf, Dn * Dn);
    cvt_bf16_kernel<<<(Dn * Dn + 255) / 256, 256, 0, stream>>>(Wv, wv_bf, Dn * Dn);
    cvt_bf16_kernel<<<(Dn * Dn + 255) / 256, 256, 0, stream>>>(Wd, wd_bf, Dn * Dn);
    cvt_bf16_kernel<<<(Hn * Dn + 255) / 256, 256, 0, stream>>>(Wcb, wcb_bf, Hn * Dn);

    // 2) projections
    dim3 gB(M / 128, Dn / 128), blk(256);
    gemm_nt<true,  false, false, false><<<gB, blk, 0, stream>>>(x_bf, wq_bf, M, Dn, Dn, nullptr, nullptr, q_f, nullptr);
    gemm_nt<false, true,  false, false><<<gB, blk, 0, stream>>>(x_bf, wk_bf, M, Dn, Dn, nullptr, nullptr, nullptr, k_bf);
    gemm_nt<false, true,  true,  false><<<gB, blk, 0, stream>>>(x_bf, wv_bf, M, Dn, Dn, bv, nullptr, nullptr, v_bf);
    gemm_nt<true,  false, false, false><<<dim3(M / 128, 1), blk, 0, stream>>>(x_bf, wcb_bf, M, Hn, Dn, nullptr, nullptr, cb_f, nullptr);

    // 3) flash attention
    attn_kernel<<<dim3(Sn / 32, Hn, Bn), 128, 0, stream>>>(q_f, k_bf, v_bf, cb_f, mixing, ctx_bf);

    // 4) output projection + bias + residual
    gemm_nt<true, false, true, true><<<gB, blk, 0, stream>>>(ctx_bf, wd_bf, M, Dn, Dn, bd, x, res_f, nullptr);

    // 5) layernorm
    ln_kernel<<<M, 256, 0, stream>>>(res_f, gamma, beta, out);
}
